// BasicTransformerBlock_22247930593741
// MI455X (gfx1250) — matrix operand order repaired and checked by reading, not compiled, not run on hardware
//
#include <hip/hip_runtime.h>
#include <hip/hip_bf16.h>

// ---------------------------------------------------------------------------
// BasicTransformerBlock forward for MI455X (gfx1250, wave32, WMMA bf16)
// ---------------------------------------------------------------------------

typedef __bf16 bf16;
typedef __attribute__((ext_vector_type(16))) __bf16 bf16x16;
typedef __attribute__((ext_vector_type(8)))  __bf16 bf16x8;
typedef __attribute__((ext_vector_type(4)))  __bf16 bf16x4;
typedef __attribute__((ext_vector_type(2)))  __bf16 bf16x2;
typedef __attribute__((ext_vector_type(8)))  float  f32x8;

#define BB      2
#define SEQ     2048
#define DMODEL  1024
#define NH      16
#define DHEAD   64
#define FF      4096
#define ROWS    (BB * SEQ)          // 4096 token rows

// ---------------------------------------------------------------------------
// WMMA helpers
// ---------------------------------------------------------------------------

__device__ __forceinline__ f32x8 wmma_bf16(bf16x16 a, bf16x16 b, f32x8 c) {
    // D = A(16x32 bf16) * B(32x16 bf16) + C(16x16 f32)
    return __builtin_amdgcn_wmma_f32_16x16x32_bf16(
        /*neg_a=*/false, a, /*neg_b=*/false, b,
        /*c_mod=*/(short)0, c, /*reuse_a=*/false, /*reuse_b=*/false);
}

// A-matrix fragment (16x32, row-major source, per ISA 7.12.2 table):
// lane = kh*16+m ; v0..v3 hold K = kh*8..kh*8+7 ; v4..v7 hold K = 16+kh*8..+7
__device__ __forceinline__ bf16x16 load_frag_a(const bf16* base, int stride) {
    const int lane = threadIdx.x & 31;
    const int m    = lane & 15;
    const int kh   = lane >> 4;
    const bf16* p  = base + m * stride + kh * 8;
    bf16x8 lo = *(const bf16x8*)(p);
    bf16x8 hi = *(const bf16x8*)(p + 16);
    bf16x16 r;
#pragma unroll
    for (int i = 0; i < 8; ++i) { r[i] = lo[i]; r[i + 8] = hi[i]; }
    return r;
}

// B-matrix fragment (32x16). Source is B^T row-major: row n holds B[:,n].
// lane = kh*16+n ; v0..v3 hold K = kh*8..kh*8+7 ; v4..v7 hold K = 16+kh*8..+7
__device__ __forceinline__ bf16x16 load_frag_b(const bf16* baseT, int stride) {
    const int lane = threadIdx.x & 31;
    const int n    = lane & 15;
    const int kh   = lane >> 4;
    const bf16* p  = baseT + n * stride + kh * 8;
    bf16x8 lo = *(const bf16x8*)(p);
    bf16x8 hi = *(const bf16x8*)(p + 16);
    bf16x16 r;
#pragma unroll
    for (int i = 0; i < 8; ++i) { r[i] = lo[i]; r[i + 8] = hi[i]; }
    return r;
}

// ---------------------------------------------------------------------------
// fp32 -> bf16 conversion (weights / activations), 4 elems per thread
// ---------------------------------------------------------------------------
__global__ __launch_bounds__(256) void k_f32_to_bf16(const float* __restrict__ in,
                                                     bf16* __restrict__ out) {
    const size_t i = (size_t)blockIdx.x * blockDim.x + threadIdx.x;
    float4 v = ((const float4*)in)[i];
    bf16x4 o;
    o[0] = (bf16)v.x; o[1] = (bf16)v.y; o[2] = (bf16)v.z; o[3] = (bf16)v.w;
    ((bf16x4*)out)[i] = o;
}

// ---------------------------------------------------------------------------
// LayerNorm over D=1024 (one block / row, 256 threads, 4 elems / thread)
// ---------------------------------------------------------------------------
__global__ __launch_bounds__(256) void k_layernorm(const float* __restrict__ x,
                                                   const float* __restrict__ gamma,
                                                   const float* __restrict__ beta,
                                                   float* __restrict__ outF,
                                                   bf16* __restrict__ outB) {
    __shared__ float red[16];
    const int row  = blockIdx.x;
    const int t    = threadIdx.x;
    const int wave = t >> 5;
    const int lane = t & 31;

    const float4 v = ((const float4*)(x + (size_t)row * DMODEL))[t];
    float s  = v.x + v.y + v.z + v.w;
    float s2 = v.x * v.x + v.y * v.y + v.z * v.z + v.w * v.w;
#pragma unroll
    for (int o = 16; o > 0; o >>= 1) {
        s  += __shfl_xor(s, o, 32);
        s2 += __shfl_xor(s2, o, 32);
    }
    if (lane == 0) { red[wave] = s; red[8 + wave] = s2; }
    __syncthreads();
    float mu = 0.f, m2 = 0.f;
#pragma unroll
    for (int i = 0; i < 8; ++i) { mu += red[i]; m2 += red[8 + i]; }
    mu *= (1.f / DMODEL);
    m2  = m2 * (1.f / DMODEL) - mu * mu;
    const float rstd = rsqrtf(m2 + 1e-5f);

    const float4 g4 = ((const float4*)gamma)[t];
    const float4 b4 = ((const float4*)beta)[t];
    float4 y;
    y.x = (v.x - mu) * rstd * g4.x + b4.x;
    y.y = (v.y - mu) * rstd * g4.y + b4.y;
    y.z = (v.z - mu) * rstd * g4.z + b4.z;
    y.w = (v.w - mu) * rstd * g4.w + b4.w;

    if (outF) ((float4*)(outF + (size_t)row * DMODEL))[t] = y;
    if (outB) {
        bf16x4 o;
        o[0] = (bf16)y.x; o[1] = (bf16)y.y; o[2] = (bf16)y.z; o[3] = (bf16)y.w;
        ((bf16x4*)(outB + (size_t)row * DMODEL))[t] = o;
    }
}

// ---------------------------------------------------------------------------
// Tiled bf16 WMMA GEMM:  C[M,N] = A[M,K] @ W[K,N]
// Block tile 128x128, K-step 32, 256 threads (8 waves, each a 32x64 patch).
// Register double-buffered: next K-tile's global loads are issued while the
// current tile's WMMAs run, hiding HBM latency with zero extra LDS.
// B tile is staged transposed with packed 2xbf16 DS stores (b32, not b16).
// Epilogue modes:
//   0: outF = acc (+bias)
//   1: QKV head scatter -> outF fp32 [B,H,N,DH] and outB bf16 same layout
//   2: outF = acc + bias + resid   (fp32, row-major; outF may alias resid)
//   3: outB = bf16(acc + bias)     (row-major)
// ---------------------------------------------------------------------------
#define TBM 128
#define TBN 128
#define TBK 32
#define LDP 40   // padded LDS stride (elems): 80B rows -> 16B-aligned chunks

__global__ __launch_bounds__(256) void k_gemm_bf16(const bf16* __restrict__ A,
                                                   const bf16* __restrict__ W,
                                                   int M, int K, int N,
                                                   const float* __restrict__ bias,
                                                   const float* resid,
                                                   float* outF,
                                                   bf16* outB,
                                                   int mode) {
    __shared__ bf16 As[TBM * LDP];   // A tile rows (padded)
    __shared__ bf16 Bs[TBN * LDP];   // W tile transposed: Bs[n][k]

    const int t    = threadIdx.x;
    const int wave = t >> 5;
    const int lane = t & 31;
    const int bm   = blockIdx.y * TBM;
    const int bn   = blockIdx.x * TBN;
    const int wm   = (wave >> 1) * 32;   // wave row offset inside tile
    const int wn   = (wave & 1) * 64;    // wave col offset inside tile

    f32x8 acc[2][4];
#pragma unroll
    for (int mi = 0; mi < 2; ++mi)
#pragma unroll
        for (int j = 0; j < 4; ++j)
#pragma unroll
            for (int e = 0; e < 8; ++e) acc[mi][j][e] = 0.f;

    // staging slices owned by this thread
    const int arow  = t >> 1;              // 0..127
    const int ahalf = (t & 1) * 16;        // 0 / 16
    const bf16* aSrc = A + (size_t)(bm + arow) * K + ahalf;

    const int kpair = (t & 15) * 2;        // 0..30 (pair of K rows)
    const int bn0   = (t >> 4) * 8;        // 0..120 (8 columns)
    const bf16* bSrc = W + (size_t)kpair * N + bn + bn0;

    // prologue: load K-tile 0 into registers
    bf16x8 ra0 = *(const bf16x8*)(aSrc);
    bf16x8 ra1 = *(const bf16x8*)(aSrc + 8);
    bf16x8 rb0 = *(const bf16x8*)(bSrc);
    bf16x8 rb1 = *(const bf16x8*)(bSrc + N);

    for (int k0 = 0; k0 < K; k0 += TBK) {
        // --- commit staged registers to LDS ---
        *(bf16x8*)(As + arow * LDP + ahalf)     = ra0;
        *(bf16x8*)(As + arow * LDP + ahalf + 8) = ra1;
#pragma unroll
        for (int i = 0; i < 8; ++i) {
            bf16x2 p; p[0] = rb0[i]; p[1] = rb1[i];
            *(bf16x2*)(Bs + (bn0 + i) * LDP + kpair) = p;   // packed b32 store
        }
        __syncthreads();

        // --- prefetch next K-tile (overlaps with WMMA below) ---
        if (k0 + TBK < K) {
            const bf16* aN = aSrc + (k0 + TBK);
            const bf16* bN = bSrc + (size_t)(k0 + TBK) * N;
            ra0 = *(const bf16x8*)(aN);
            ra1 = *(const bf16x8*)(aN + 8);
            rb0 = *(const bf16x8*)(bN);
            rb1 = *(const bf16x8*)(bN + N);
        }

        // --- compute ---
        bf16x16 af[2], bfr[4];
        af[0] = load_frag_a(As + (wm +  0) * LDP, LDP);
        af[1] = load_frag_a(As + (wm + 16) * LDP, LDP);
#pragma unroll
        for (int j = 0; j < 4; ++j)
            bfr[j] = load_frag_b(Bs + (wn + j * 16) * LDP, LDP);
#pragma unroll
        for (int mi = 0; mi < 2; ++mi)
#pragma unroll
            for (int j = 0; j < 4; ++j)
                acc[mi][j] = wmma_bf16(af[mi], bfr[j], acc[mi][j]);
        __syncthreads();
    }

    // --- epilogue (C layout: n = lane&15, m = vgpr + 8*(lane>>4)) ---
    const int cn  = lane & 15;
    const int mhi = (lane >> 4) * 8;
#pragma unroll
    for (int mi = 0; mi < 2; ++mi) {
#pragma unroll
        for (int j = 0; j < 4; ++j) {
#pragma unroll
            for (int e = 0; e < 8; ++e) {
                const int gm = bm + wm + mi * 16 + mhi + e;
                const int gn = bn + wn + j * 16 + cn;
                float v = acc[mi][j][e];
                if (mode == 0) {
                    if (bias) v += bias[gn];
                    outF[(size_t)gm * N + gn] = v;
                } else if (mode == 1) {
                    const int bb = gm >> 11;      // row / SEQ
                    const int nn = gm & (SEQ - 1);
                    const int hh = gn >> 6;       // col / DHEAD
                    const int dd = gn & (DHEAD - 1);
                    const size_t idx =
                        ((size_t)(bb * NH + hh) * SEQ + nn) * DHEAD + dd;
                    outF[idx] = v;
                    outB[idx] = (bf16)v;
                } else if (mode == 2) {
                    v += bias[gn] + resid[(size_t)gm * N + gn];
                    outF[(size_t)gm * N + gn] = v;
                } else { // mode 3
                    v += bias[gn];
                    outB[(size_t)gm * N + gn] = (bf16)v;
                }
            }
        }
    }
}

// ---------------------------------------------------------------------------
// Attention: one workgroup per (bh, 16-row tile of queries).
// S row-block (16 x 2048 fp32) lives in LDS (320KB WGP LDS makes the exact
// softmax layout feasible). QK^T B-fragments come straight from K's row-major
// global rows (K^T column == K row). P*V splits the K range across wave
// halves so all 8 waves drive the matrix units; partials reduced via LDS.
// ---------------------------------------------------------------------------
#define SM_S     0
#define SM_P     (SEQ * 16 * 4)                       // + 16x2048 bf16
#define SM_Q     (SM_P + SEQ * 16 * 2)                // + 16x64 bf16
#define SM_V     (SM_Q + 16 * DHEAD * 2)              // + 64xLDP bf16
#define SM_PART  (SM_V + 64 * LDP * 2)                // + 4x256 f32
#define ATTN_SMEM (SM_PART + 4 * 256 * 4)

__global__ __launch_bounds__(256) void k_attention(const bf16* __restrict__ Q,
                                                   const bf16* __restrict__ Kb,
                                                   const bf16* __restrict__ Vb,
                                                   float* __restrict__ interOut,
                                                   bf16* __restrict__ attnRow) {
    extern __shared__ char smem[];
    float* S     = (float*)(smem + SM_S);     // 16 x 2048 f32
    bf16*  P     = (bf16*)(smem + SM_P);      // 16 x 2048 bf16
    bf16*  Qs    = (bf16*)(smem + SM_Q);      // 16 x 64
    bf16*  Vs    = (bf16*)(smem + SM_V);      // 64 x LDP (transposed V chunk)
    float* Spart = (float*)(smem + SM_PART);  // 4 x 16 x 16 partials

    const int t    = threadIdx.x;
    const int wave = t >> 5;
    const int lane = t & 31;
    const int bh   = blockIdx.y;   // 0..B*H-1
    const int it   = blockIdx.x;   // 0..SEQ/16-1

    const bf16* qp = Q  + ((size_t)bh * SEQ + it * 16) * DHEAD;
    const bf16* kp = Kb + (size_t)bh * SEQ * DHEAD;
    const bf16* vp = Vb + (size_t)bh * SEQ * DHEAD;

    // stage Q tile (16x64)
    if (t < 128) {
        const int row = t >> 3;
        const int c   = (t & 7) * 8;
        *(bf16x8*)(Qs + row * DHEAD + c) = *(const bf16x8*)(qp + row * DHEAD + c);
    }
    __syncthreads();

    // ---- phase 1: S = (Q K^T) * scale ----
    bf16x16 aq0 = load_frag_a(Qs, DHEAD);
    bf16x16 aq1 = load_frag_a(Qs + 32, DHEAD);
    const int cn  = lane & 15;
    const int mhi = (lane >> 4) * 8;
    for (int jt = 0; jt < 16; ++jt) {
        const int j0 = wave * 256 + jt * 16;
        f32x8 acc;
#pragma unroll
        for (int e = 0; e < 8; ++e) acc[e] = 0.f;
        bf16x16 b0 = load_frag_b(kp + (size_t)j0 * DHEAD, DHEAD);
        bf16x16 b1 = load_frag_b(kp + (size_t)j0 * DHEAD + 32, DHEAD);
        acc = wmma_bf16(aq0, b0, acc);
        acc = wmma_bf16(aq1, b1, acc);
#pragma unroll
        for (int e = 0; e < 8; ++e)
            S[(mhi + e) * SEQ + j0 + cn] = acc[e] * 0.125f;   // DH^-0.5
    }
    __syncthreads();

    // ---- phase 2: exact softmax per row (wave w owns rows 2w, 2w+1) ----
    for (int rr = 0; rr < 2; ++rr) {
        const int r  = wave * 2 + rr;
        float* sr = S + (size_t)r * SEQ;
        float mx = -3.0e38f;
        for (int c = lane * 2; c < SEQ; c += 64) {
            const float2 v = *(const float2*)(sr + c);
            mx = fmaxf(mx, fmaxf(v.x, v.y));
        }
#pragma unroll
        for (int o = 16; o > 0; o >>= 1) mx = fmaxf(mx, __shfl_xor(mx, o, 32));
        float sum = 0.f;
        for (int c = lane * 2; c < SEQ; c += 64) {
            float2 v = *(const float2*)(sr + c);
            v.x = __expf(v.x - mx);
            v.y = __expf(v.y - mx);
            *(float2*)(sr + c) = v;
            sum += v.x + v.y;
        }
#pragma unroll
        for (int o = 16; o > 0; o >>= 1) sum += __shfl_xor(sum, o, 32);
        const float inv = 1.f / sum;
        for (int c = lane * 2; c < SEQ; c += 64) {
            const float2 v = *(const float2*)(sr + c);
            bf16x2 p;
            p[0] = (bf16)(v.x * inv);
            p[1] = (bf16)(v.y * inv);
            *(bf16x2*)(P + (size_t)r * SEQ + c) = p;
        }
    }
    __syncthreads();

    // ---- phase 3: O = P V ; wave = kh2*4 + ct (col tile ct, K-half kh2) ----
    const int ct  = wave & 3;
    const int kh2 = wave >> 2;
    f32x8 acc;
#pragma unroll
    for (int e = 0; e < 8; ++e) acc[e] = 0.f;
    const int vkp = (t & 31) * 2;    // K pair inside 64-row chunk
    const int vn0 = (t >> 5) * 8;    // 8 columns
    for (int kt = 0; kt < SEQ / 64; ++kt) {
        // all waves stage a 64x64 V chunk transposed into Vs[n][k]
        const bf16* vrow = vp + (size_t)(kt * 64 + vkp) * DHEAD + vn0;
        bf16x8 v0 = *(const bf16x8*)(vrow);
        bf16x8 v1 = *(const bf16x8*)(vrow + DHEAD);
#pragma unroll
        for (int i = 0; i < 8; ++i) {
            bf16x2 p; p[0] = v0[i]; p[1] = v1[i];
            *(bf16x2*)(Vs + (vn0 + i) * LDP + vkp) = p;     // packed b32 store
        }
        __syncthreads();
        bf16x16 pa = load_frag_a(P + kt * 64 + kh2 * 32, SEQ);
        bf16x16 vb = load_frag_b(Vs + (ct * 16) * LDP + kh2 * 32, LDP);
        acc = wmma_bf16(pa, vb, acc);
        __syncthreads();
    }

    // reduce the two K-halves through LDS
    if (kh2 == 1) {
#pragma unroll
        for (int e = 0; e < 8; ++e)
            Spart[ct * 256 + (mhi + e) * 16 + cn] = acc[e];
    }
    __syncthreads();
    if (kh2 == 0) {
        const int b = bh >> 4;
        const int h = bh & (NH - 1);
#pragma unroll
        for (int e = 0; e < 8; ++e) {
            const int row = it * 16 + mhi + e;
            const int d   = ct * 16 + cn;
            const float v = acc[e] + Spart[ct * 256 + (mhi + e) * 16 + cn];
            interOut[((size_t)bh * SEQ + row) * DHEAD + d] = v;              // [B,H,N,DH]
            attnRow[((size_t)(b * SEQ + row)) * (NH * DHEAD) + h * DHEAD + d] = (bf16)v;
        }
    }
}

// ---------------------------------------------------------------------------
// GEGLU: gated[r,c] = h[r,c] * gelu(h[r, FF+c])   (exact erf gelu)
// ---------------------------------------------------------------------------
__global__ __launch_bounds__(256) void k_geglu(const bf16* __restrict__ h,
                                               bf16* __restrict__ out) {
    const size_t i = (size_t)blockIdx.x * 256 + threadIdx.x;   // over ROWS*FF
    const size_t r = i >> 12;          // / FF
    const size_t c = i & (FF - 1);
    const float a = (float)h[r * (2 * FF) + c];
    const float g = (float)h[r * (2 * FF) + FF + c];
    const float ge = 0.5f * g * (1.0f + erff(g * 0.70710678118654752f));
    out[i] = (bf16)(a * ge);
}

// ---------------------------------------------------------------------------
// Host-side orchestration
// ---------------------------------------------------------------------------
extern "C" void kernel_launch(void* const* d_in, const int* in_sizes, int n_in,
                              void* d_out, int out_size, void* d_ws, size_t ws_size,
                              hipStream_t stream) {
    (void)in_sizes; (void)n_in; (void)out_size; (void)ws_size;

    const float* x    = (const float*)d_in[0];
    const float* Wq   = (const float*)d_in[1];
    const float* Wk   = (const float*)d_in[2];
    const float* Wv   = (const float*)d_in[3];
    const float* Wo   = (const float*)d_in[4];
    const float* bo   = (const float*)d_in[5];
    const float* ln1g = (const float*)d_in[6];
    const float* ln1b = (const float*)d_in[7];
    const float* ln3g = (const float*)d_in[8];
    const float* ln3b = (const float*)d_in[9];
    const float* Wff1 = (const float*)d_in[10];
    const float* bff1 = (const float*)d_in[11];
    const float* Wff2 = (const float*)d_in[12];
    const float* bff2 = (const float*)d_in[13];

    // outputs, concatenated flat in return order
    float* out_x     = (float*)d_out;                         // (B,N,D)
    float* out_q     = out_x + (size_t)ROWS * DMODEL;         // (B*H,N,DH)
    float* out_k     = out_q + (size_t)ROWS * DMODEL;
    float* out_v     = out_k + (size_t)ROWS * DMODEL;
    float* out_inter = out_v + (size_t)ROWS * DMODEL;         // (B,H,N,DH)
    float* out_xf    = out_inter + (size_t)ROWS * DMODEL;     // (B,N,D)

    // workspace carve-up (bf16 scratch)
    char* w = (char*)d_ws;
    bf16* xf_bf   = (bf16*)w; w += (size_t)ROWS * DMODEL * 2;        // ln1 out
    bf16* wq_bf   = (bf16*)w; w += (size_t)DMODEL * DMODEL * 2;
    bf16* wk_bf   = (bf16*)w; w += (size_t)DMODEL * DMODEL * 2;
    bf16* wv_bf   = (bf16*)w; w += (size_t)DMODEL * DMODEL * 2;
    bf16* wo_bf   = (bf16*)w; w += (size_t)DMODEL * DMODEL * 2;
    bf16* wff1_bf = (bf16*)w; w += (size_t)DMODEL * 2 * FF * 2;
    bf16* wff2_bf = (bf16*)w; w += (size_t)FF * DMODEL * 2;
    bf16* q_bf    = (bf16*)w; w += (size_t)ROWS * DMODEL * 2;        // [B*H,N,DH]
    bf16* k_bf    = (bf16*)w; w += (size_t)ROWS * DMODEL * 2;
    bf16* v_bf    = (bf16*)w; w += (size_t)ROWS * DMODEL * 2;
    bf16* attn_bf = (bf16*)w; w += (size_t)ROWS * DMODEL * 2;        // (B,N,INNER)
    bf16* x1_bf   = (bf16*)w; w += (size_t)ROWS * DMODEL * 2;        // ln3 out
    bf16* h_bf    = (bf16*)w; w += (size_t)ROWS * 2 * FF * 2;        // FF1 out
    bf16* gate_bf = (bf16*)w; w += (size_t)ROWS * FF * 2;            // geglu out

    // 1) weights -> bf16 (1024 elems per block: 256 thr x 4)
    k_f32_to_bf16<<<(DMODEL * DMODEL) / 1024, 256, 0, stream>>>(Wq, wq_bf);
    k_f32_to_bf16<<<(DMODEL * DMODEL) / 1024, 256, 0, stream>>>(Wk, wk_bf);
    k_f32_to_bf16<<<(DMODEL * DMODEL) / 1024, 256, 0, stream>>>(Wv, wv_bf);
    k_f32_to_bf16<<<(DMODEL * DMODEL) / 1024, 256, 0, stream>>>(Wo, wo_bf);
    k_f32_to_bf16<<<(DMODEL * 2 * FF) / 1024, 256, 0, stream>>>(Wff1, wff1_bf);
    k_f32_to_bf16<<<(FF * DMODEL) / 1024, 256, 0, stream>>>(Wff2, wff2_bf);

    // 2) LayerNorm1 -> x_features (f32 output) + bf16 copy
    k_layernorm<<<ROWS, 256, 0, stream>>>(x, ln1g, ln1b, out_xf, xf_bf);

    // 3) q/k/v projections (mode 1: head scatter, f32 + bf16)
    dim3 gProj(DMODEL / TBN, ROWS / TBM);
    k_gemm_bf16<<<gProj, 256, 0, stream>>>(xf_bf, wq_bf, ROWS, DMODEL, DMODEL,
                                           nullptr, nullptr, out_q, q_bf, 1);
    k_gemm_bf16<<<gProj, 256, 0, stream>>>(xf_bf, wk_bf, ROWS, DMODEL, DMODEL,
                                           nullptr, nullptr, out_k, k_bf, 1);
    k_gemm_bf16<<<gProj, 256, 0, stream>>>(xf_bf, wv_bf, ROWS, DMODEL, DMODEL,
                                           nullptr, nullptr, out_v, v_bf, 1);

    // 4) attention (exact softmax, S block in LDS)
    dim3 gAttn(SEQ / 16, BB * NH);
    k_attention<<<gAttn, 256, ATTN_SMEM, stream>>>(q_bf, k_bf, v_bf,
                                                   out_inter, attn_bf);

    // 5) Wo projection + bias + residual(x) -> out_x (f32)
    k_gemm_bf16<<<gProj, 256, 0, stream>>>(attn_bf, wo_bf, ROWS, DMODEL, DMODEL,
                                           bo, x, out_x, nullptr, 2);

    // 6) LayerNorm3 -> bf16 only
    k_layernorm<<<ROWS, 256, 0, stream>>>(out_x, ln3g, ln3b, nullptr, x1_bf);

    // 7) FF1: (4096 x 1024) @ (1024 x 8192) + bff1 -> bf16 h
    dim3 gFF1((2 * FF) / TBN, ROWS / TBM);
    k_gemm_bf16<<<gFF1, 256, 0, stream>>>(x1_bf, wff1_bf, ROWS, DMODEL, 2 * FF,
                                          bff1, nullptr, nullptr, h_bf, 3);

    // 8) GEGLU: a * gelu(gate)
    k_geglu<<<((size_t)ROWS * FF) / 256, 256, 0, stream>>>(h_bf, gate_bf);

    // 9) FF2: (4096 x 4096) @ (4096 x 1024) + bff2 + residual(out_x) -> out_x
    dim3 gFF2(DMODEL / TBN, ROWS / TBM);
    k_gemm_bf16<<<gFF2, 256, 0, stream>>>(gate_bf, wff2_bf, ROWS, FF, DMODEL,
                                          bff2, out_x, out_x, nullptr, 2);
}
